// EuclideanCodebook_31619549233272
// MI455X (gfx1250) — matrix operand order repaired and checked by reading, not compiled, not run on hardware
//
#include <hip/hip_runtime.h>

// Problem sizes (fixed by the reference)
constexpr int NROWS = 4 * 4096;   // B*T = 16384 query rows
constexpr int DIM   = 512;        // feature dim
constexpr int KCB   = 8192;       // codebook size

// Tiling
constexpr int MB  = 64;           // query rows per block (4 M-waves * 16)
constexpr int NB  = 32;           // codebook cols per panel (2 N-waves * 16)
constexpr int STR = DIM + 8;      // padded LDS row stride in halfs (16B-aligned,
                                  // row stride = 4 banks -> conflict-free B reads)

typedef __bf16 v8bf  __attribute__((ext_vector_type(8)));
typedef __bf16 v16bf __attribute__((ext_vector_type(16)));
typedef float  v8f   __attribute__((ext_vector_type(8)));

__device__ __forceinline__ unsigned short f2bf(float f) {
  unsigned int u = __builtin_bit_cast(unsigned int, f);
  u += 0x7FFFu + ((u >> 16) & 1u);          // round-to-nearest-even
  return (unsigned short)(u >> 16);
}
__device__ __forceinline__ float bf2f(unsigned short h) {
  unsigned int u = ((unsigned int)h) << 16;
  return __builtin_bit_cast(float, u);
}

__device__ __forceinline__ void wait_async0() {
#if __has_builtin(__builtin_amdgcn_s_wait_asynccnt)
  __builtin_amdgcn_s_wait_asynccnt(0);
#else
  asm volatile("s_wait_asynccnt 0" ::: "memory");
#endif
}

// Async global->LDS copy of 16 bytes per lane (gfx1250, ASYNCcnt-tracked).
__device__ __forceinline__ void async_copy_b128(unsigned ldsAddr, unsigned gOff,
                                                unsigned long long gBase) {
  asm volatile("global_load_async_to_lds_b128 %0, %1, %2"
               :: "v"(ldsAddr), "v"(gOff), "s"(gBase) : "memory");
}

// ---------------------------------------------------------------------------
// Kernel 1: codebook prep — bf16 hi/lo split + 0.5*||e||^2
// ---------------------------------------------------------------------------
__global__ __launch_bounds__(256) void prep_e_kernel(
    const float* __restrict__ embed,
    unsigned short* __restrict__ e_hi,
    unsigned short* __restrict__ e_lo,
    float* __restrict__ hesq)
{
  __shared__ float red[256];
  const int k = blockIdx.x;
  const int tid = threadIdx.x;
  float ss = 0.f;
  for (int c = tid; c < DIM; c += 256) {
    float v = embed[(size_t)k * DIM + c];
    unsigned short hi = f2bf(v);
    float rem = v - bf2f(hi);
    e_hi[(size_t)k * DIM + c] = hi;
    e_lo[(size_t)k * DIM + c] = f2bf(rem);
    ss += v * v;
  }
  red[tid] = ss;
  __syncthreads();
  for (int s = 128; s > 0; s >>= 1) {
    if (tid < s) red[tid] += red[tid + s];
    __syncthreads();
  }
  if (tid == 0) hesq[k] = 0.5f * red[0];
}

// ---------------------------------------------------------------------------
// Kernel 2: fused (X @ E^T) GEMM + running argmax of (x.e - 0.5*||e||^2)
// 256 threads = 8 waves: wave w -> mw = w>>1 (M subtile), nw = w&1 (N subtile).
// A (hi+lo, full D) lives in VGPRs for the whole kernel; LDS holds only the
// double-buffered E panel, filled by async global->LDS copies overlapping the
// WMMAs. Steady-state LDS reads are B-only: 4 x ds_load_b128 per 3 WMMAs.
// ---------------------------------------------------------------------------
__global__ __launch_bounds__(256) void vq_argmax_kernel(
    const float* __restrict__ x,
    const unsigned short* __restrict__ e_hi,
    const unsigned short* __restrict__ e_lo,
    const float* __restrict__ hesq,
    int* __restrict__ best_out)
{
  // LDS region: reused. Phase 1: X tile (hi | lo), each MB*STR halfs.
  // Phase 2: E panels, 2 bufs * (hi | lo) * NB*STR halfs. Same total size.
  extern __shared__ unsigned short smem[];
  unsigned short* Xh   = smem;                 // MB * STR
  unsigned short* Xl   = Xh + MB * STR;        // MB * STR
  unsigned short* Epan = smem;                 // 2 * 2 * NB * STR (== 2*MB*STR)

  const int tid  = threadIdx.x;
  const int lane = tid & 31;
  const int w    = tid >> 5;                 // 0..7
  const int mw   = w >> 1;                   // 0..3
  const int nw   = w & 1;                    // 0..1
  const int row0 = blockIdx.x * MB;

  const unsigned long long ehBase = (unsigned long long)(uintptr_t)e_hi;
  const unsigned long long elBase = (unsigned long long)(uintptr_t)e_lo;

  // Async-stage one 32-column codebook panel (hi+lo) into LDS buffer `buf`.
  auto stage_panel = [&](int n0, int buf) {
    unsigned short* Ehb = Epan + buf * (2 * NB * STR);
    unsigned short* Elb = Ehb + NB * STR;
    // NB*DIM/8 = 2048 chunks of 8 halfs (16B); 256 threads -> 8 chunks each
    for (int i = tid; i < NB * DIM / 8; i += 256) {
      int r = i >> 6;                  // DIM/8 = 64 chunks per row
      int c = (i & 63) << 3;
      unsigned gOff = (unsigned)((((unsigned)(n0 + r)) * DIM + c) * 2u);
      async_copy_b128((unsigned)(uintptr_t)(Ehb + r * STR + c), gOff, ehBase);
      async_copy_b128((unsigned)(uintptr_t)(Elb + r * STR + c), gOff, elBase);
    }
  };

  // Phase 1a: stage + split the X tile into LDS (float4 in, b64 packed out)
  for (int i = tid; i < MB * DIM / 4; i += 256) {
    int r = i >> 7, c = (i & 127) << 2;
    const float4 v = *(const float4*)(x + (size_t)(row0 + r) * DIM + c);
    unsigned short h0 = f2bf(v.x), h1 = f2bf(v.y), h2 = f2bf(v.z), h3 = f2bf(v.w);
    unsigned short l0 = f2bf(v.x - bf2f(h0)), l1 = f2bf(v.y - bf2f(h1));
    unsigned short l2 = f2bf(v.z - bf2f(h2)), l3 = f2bf(v.w - bf2f(h3));
    uint2 hp, lp;
    hp.x = (unsigned)h0 | ((unsigned)h1 << 16);
    hp.y = (unsigned)h2 | ((unsigned)h3 << 16);
    lp.x = (unsigned)l0 | ((unsigned)l1 << 16);
    lp.y = (unsigned)l2 | ((unsigned)l3 << 16);
    *(uint2*)(Xh + r * STR + c) = hp;
    *(uint2*)(Xl + r * STR + c) = lp;
  }
  __syncthreads();

  // Per-lane WMMA operand addressing (A 16x32 / B 32x16, same contiguous k = 16*half + s order in both)
  const int arow = mw * 16 + (lane & 15);        // query row within tile
  const int ak   = (lane >> 4) * 16;             // A K-base: +16 for hi half-wave
  const int brow = nw * 16 + (lane & 15);        // codebook col within panel
  const int bk   = (lane >> 4) * 16;             // B K-base: +16 for hi half-wave
  const int ncol = nw * 16 + (lane & 15);        // C/D column for this lane

  // Phase 1b: hoist this wave's A operands (hi+lo, all 16 d-steps) into VGPRs
  auto load_A = [&](const unsigned short* p, int d0) -> v16bf {
    v8bf g0 = *(const v8bf*)(p + d0 + ak);
    v8bf g1 = *(const v8bf*)(p + d0 + ak + 8);
    return __builtin_shufflevector(g0, g1, 0,1,2,3,4,5,6,7,8,9,10,11,12,13,14,15);
  };
  const unsigned short* xh = Xh + arow * STR;
  const unsigned short* xl = Xl + arow * STR;
  v16bf AhR[16], AlR[16];
#pragma unroll
  for (int t = 0; t < 16; ++t) {
    AhR[t] = load_A(xh, t * 32);
    AlR[t] = load_A(xl, t * 32);
  }
  __syncthreads();          // everyone done reading X; LDS becomes E panels

  // Phase 2: stream the codebook, double-buffered async panels
  stage_panel(0, 0);

  float bestVal[8];
  int   bestIdx[8];
#pragma unroll
  for (int v = 0; v < 8; ++v) { bestVal[v] = -3.4e38f; bestIdx[v] = 0; }

  for (int n0 = 0; n0 < KCB; n0 += NB) {
    const int buf = (n0 / NB) & 1;
    wait_async0();        // my async copies for this panel have landed
    __syncthreads();      // everyone's have landed

    // Prefetch next panel into the other buffer (overlaps the WMMAs below)
    if (n0 + NB < KCB) stage_panel(n0 + NB, buf ^ 1);

    unsigned short* Ehb = Epan + buf * (2 * NB * STR);
    unsigned short* Elb = Ehb + NB * STR;
    const unsigned short* eh = Ehb + brow * STR;
    const unsigned short* el = Elb + brow * STR;

    v8f a0 = {0.f,0.f,0.f,0.f,0.f,0.f,0.f,0.f};
    v8f a1 = {0.f,0.f,0.f,0.f,0.f,0.f,0.f,0.f};
    v8f a2 = {0.f,0.f,0.f,0.f,0.f,0.f,0.f,0.f};

#pragma unroll
    for (int t = 0; t < 16; ++t) {
      v16bf Bh = *(const v16bf*)(eh + t * 32 + bk);
      v16bf Bl = *(const v16bf*)(el + t * 32 + bk);
      // split-bf16 fp32 emulation on 3 independent accumulator chains
      a0 = __builtin_amdgcn_wmma_f32_16x16x32_bf16(false, AhR[t], false, Bh, (short)0, a0, false, false);
      a1 = __builtin_amdgcn_wmma_f32_16x16x32_bf16(false, AhR[t], false, Bl, (short)0, a1, false, false);
      a2 = __builtin_amdgcn_wmma_f32_16x16x32_bf16(false, AlR[t], false, Bh, (short)0, a2, false, false);
    }
    v8f acc = a0 + a1 + a2;

    // Fused epilogue: score = x.e - 0.5*||e||^2 ; running argmax in registers
    int   n = n0 + ncol;
    float h = hesq[n];
#pragma unroll
    for (int v = 0; v < 8; ++v) {
      float s = acc[v] - h;
      if (s > bestVal[v]) { bestVal[v] = s; bestIdx[v] = n; }
    }
  }

  // Butterfly argmax across the 16 lanes of each half-wave (cols of a row)
#pragma unroll
  for (int m = 1; m < 16; m <<= 1) {
#pragma unroll
    for (int v = 0; v < 8; ++v) {
      float ov = __shfl_xor(bestVal[v], m, 32);
      int   oi = __shfl_xor(bestIdx[v], m, 32);
      if (ov > bestVal[v] || (ov == bestVal[v] && oi < bestIdx[v])) {
        bestVal[v] = ov; bestIdx[v] = oi;
      }
    }
  }

  // Combine the two N-waves covering the same rows via LDS, write indices
  __syncthreads();
  float* sval = (float*)smem;                 // [8 waves][16 rows]
  int*   sidx = (int*)(sval + 8 * 16);
  if ((lane & 15) == 0) {
    int half = lane >> 4;                     // lanes 0-15: rows v; 16-31: rows 8+v
#pragma unroll
    for (int v = 0; v < 8; ++v) {
      sval[w * 16 + half * 8 + v] = bestVal[v];
      sidx[w * 16 + half * 8 + v] = bestIdx[v];
    }
  }
  __syncthreads();
  if (tid < 64) {
    int mwi = tid >> 4, ri = tid & 15;
    int w0 = mwi * 2, w1 = w0 + 1;
    float v0 = sval[w0 * 16 + ri], v1 = sval[w1 * 16 + ri];
    int   i0 = sidx[w0 * 16 + ri], i1 = sidx[w1 * 16 + ri];
    int bi = (v1 > v0 || (v1 == v0 && i1 < i0)) ? i1 : i0;
    best_out[row0 + mwi * 16 + ri] = bi;
  }
}

// ---------------------------------------------------------------------------
// Kernel 3: gather quantize = embed[ind] and emit indices (as float) to tail
// ---------------------------------------------------------------------------
__global__ __launch_bounds__(128) void gather_kernel(
    const float* __restrict__ embed,
    const int* __restrict__ idx,
    float* __restrict__ out)
{
  const int r  = blockIdx.x;
  const int id = idx[r];
  const float4* src = (const float4*)(embed + (size_t)id * DIM);
  float4*       dst = (float4*)(out + (size_t)r * DIM);
  dst[threadIdx.x] = src[threadIdx.x];        // 128 threads * float4 = 512 floats
  if (threadIdx.x == 0) out[(size_t)NROWS * DIM + r] = (float)id;
}

// ---------------------------------------------------------------------------
extern "C" void kernel_launch(void* const* d_in, const int* in_sizes, int n_in,
                              void* d_out, int out_size, void* d_ws, size_t ws_size,
                              hipStream_t stream) {
  (void)in_sizes; (void)n_in; (void)out_size; (void)ws_size;
  const float* x     = (const float*)d_in[0];
  const float* embed = (const float*)d_in[1];
  float*       out   = (float*)d_out;

  // Workspace layout: e_hi (8MB) | e_lo (8MB) | hesq (32KB) | indices (64KB)
  unsigned short* e_hi = (unsigned short*)d_ws;
  unsigned short* e_lo = e_hi + (size_t)KCB * DIM;
  float* hesq  = (float*)(e_lo + (size_t)KCB * DIM);
  int*   idxbf = (int*)(hesq + KCB);

  prep_e_kernel<<<KCB, 256, 0, stream>>>(embed, e_hi, e_lo, hesq);

  // LDS: max(X hi/lo, double-buffered E panel) = 2*MB*STR halfs (~130 KB)
  size_t ldsBytes = (size_t)(2 * MB * STR) * sizeof(unsigned short);
  vq_argmax_kernel<<<NROWS / MB, 256, ldsBytes, stream>>>(x, e_hi, e_lo, hesq, idxbf);

  gather_kernel<<<NROWS, 128, 0, stream>>>(embed, idxbf, out);
}
